// AvaAttention_36249523978775
// MI455X (gfx1250) — matrix operand order repaired and checked by reading, not compiled, not run on hardware
//
#include <hip/hip_runtime.h>

#define T_SEQ 2048
#define HIDN  2048
#define NH    16
#define NKV   8
#define HD    128
#define ATT_SCALE 0.08838834764831845f

typedef __attribute__((ext_vector_type(16))) __bf16 v16bf;
typedef __attribute__((ext_vector_type(8)))  __bf16 v8bf;
typedef __attribute__((ext_vector_type(8)))  float  v8f;

static __device__ __forceinline__ v8f vzero() {
  v8f z;
#pragma unroll
  for (int i = 0; i < 8; i++) z[i] = 0.0f;
  return z;
}

static __device__ __forceinline__ v16bf cat8(v8bf lo, v8bf hi) {
  return __builtin_shufflevector(lo, hi, 0, 1, 2, 3, 4, 5, 6, 7,
                                 8, 9, 10, 11, 12, 13, 14, 15);
}

static __device__ __forceinline__ v8f wmma_bf16(v16bf a, v16bf b, v8f c) {
  return __builtin_amdgcn_wmma_f32_16x16x32_bf16(false, a, false, b,
                                                 (short)0, c, false, false);
}

// A-fragment (16x32 bf16): per-lane row = lane&15; K elems are 16 contiguous
// elements at 16*laneHi (two 16B chunks at 16*laneHi and 16*laneHi + 8).
static __device__ __forceinline__ v16bf load_afrag(const __bf16* rowp, int laneHi) {
  v8bf lo = *(const v8bf*)(rowp + (laneHi << 4));
  v8bf hi = *(const v8bf*)(rowp + (laneHi << 4) + 8);
  return cat8(lo, hi);
}

// ---------------------------------------------------------------------------
// Kernel 1: qkv = hidden @ w_qkv^T  (M=4096, N=4096, K=2048), fused RoPE.
// Block tile 64(M) x 128(N), 8 waves in 2x4, each wave 32x32 (4 WMMA/K-step).
// Double-buffered LDS staging (one barrier per K-step).
// Q -> qb[b][h][t][d], K -> kb[b][hk][t][d], V -> vt[b][hk][d][t] (transposed).
// ---------------------------------------------------------------------------
__global__ __launch_bounds__(256) void qkv_rope_kernel(
    const float* __restrict__ hidden, const float* __restrict__ cosT,
    const float* __restrict__ sinT, const float* __restrict__ wqkv,
    void* qb_, void* kb_, void* vt_) {
  __bf16* __restrict__ qb = (__bf16*)qb_;
  __bf16* __restrict__ kb = (__bf16*)kb_;
  __bf16* __restrict__ vt = (__bf16*)vt_;
  __shared__ __align__(32) __bf16 As[2][64][32];
  __shared__ __align__(32) __bf16 Bs[2][128][32];
  __shared__ float Cs[64][128];

  const int tid = threadIdx.x;
  const int lane = tid & 31, wid = tid >> 5;
  const int laneHi = lane >> 4, laneLo = lane & 15;
  const int nb = blockIdx.x, mb = blockIdx.y;
  const int m0 = mb * 64, n0 = nb * 128;
  const int wm = wid >> 2, wn = wid & 3;  // 2 x 4 wave grid

  v8f acc00 = vzero(), acc01 = vzero(), acc10 = vzero(), acc11 = vzero();

  const int ldA_row = tid >> 2, ldA_cg = (tid & 3) << 3;  // 64 x 32
  const int ldB_row = tid >> 1, ldB_cg = (tid & 1) << 4;  // 128 x 32

  auto stage = [&](int buf, int k0) {
    {
      const float4* src =
          (const float4*)(hidden + (size_t)(m0 + ldA_row) * HIDN + k0 + ldA_cg);
      float4 f0 = src[0], f1 = src[1];
      __bf16* dst = &As[buf][ldA_row][ldA_cg];
      dst[0] = (__bf16)f0.x; dst[1] = (__bf16)f0.y;
      dst[2] = (__bf16)f0.z; dst[3] = (__bf16)f0.w;
      dst[4] = (__bf16)f1.x; dst[5] = (__bf16)f1.y;
      dst[6] = (__bf16)f1.z; dst[7] = (__bf16)f1.w;
    }
    {
      const float4* src =
          (const float4*)(wqkv + (size_t)(n0 + ldB_row) * HIDN + k0 + ldB_cg);
      __bf16* dst = &Bs[buf][ldB_row][ldB_cg];
#pragma unroll
      for (int j = 0; j < 4; j++) {
        float4 f = src[j];
        dst[4 * j + 0] = (__bf16)f.x; dst[4 * j + 1] = (__bf16)f.y;
        dst[4 * j + 2] = (__bf16)f.z; dst[4 * j + 3] = (__bf16)f.w;
      }
    }
  };

  stage(0, 0);
  __syncthreads();

  for (int k0 = 0; k0 < HIDN; k0 += 32) {
    const int cur = (k0 >> 5) & 1;
    if (k0 + 32 < HIDN) {
      stage(cur ^ 1, k0 + 32);
      if (k0 + 64 < HIDN) {  // global_prefetch_b8 two tiles ahead
        __builtin_prefetch(hidden + (size_t)(m0 + ldA_row) * HIDN + k0 + 64 + ldA_cg, 0, 3);
        __builtin_prefetch(wqkv + (size_t)(n0 + ldB_row) * HIDN + k0 + 64 + ldB_cg, 0, 3);
      }
    }
    v16bf af0 = load_afrag(&As[cur][wm * 32 + laneLo][0], laneHi);
    v16bf af1 = load_afrag(&As[cur][wm * 32 + 16 + laneLo][0], laneHi);
    v16bf bf0 = *(const v16bf*)(&Bs[cur][wn * 32 + laneLo][laneHi << 4]);
    v16bf bf1 = *(const v16bf*)(&Bs[cur][wn * 32 + 16 + laneLo][laneHi << 4]);

    acc00 = wmma_bf16(af0, bf0, acc00);
    acc01 = wmma_bf16(af0, bf1, acc01);
    acc10 = wmma_bf16(af1, bf0, acc10);
    acc11 = wmma_bf16(af1, bf1, acc11);
    __syncthreads();
  }

  // D layout: row = r + 8*laneHi, col = laneLo
#pragma unroll
  for (int r = 0; r < 8; r++) {
    const int mA = wm * 32 + r + (laneHi << 3);
    const int nA = wn * 32 + laneLo;
    Cs[mA][nA] = acc00[r];
    Cs[mA][nA + 16] = acc01[r];
    Cs[mA + 16][nA] = acc10[r];
    Cs[mA + 16][nA + 16] = acc11[r];
  }
  __syncthreads();

  // Epilogue: RoPE + scatter. 256 threads cover 64 rows x 64 pair-columns (16 per thread).
  const int row = tid >> 2;
  const int dbase = (tid & 3) << 4;
  const int bt = m0 + row;
  const int b = bt >> 11;  // T_SEQ == 2048
  const int t = bt & (T_SEQ - 1);

  if (nb < 16) {  // Q head nb
    __bf16* dst = qb + (((size_t)b * NH + nb) * T_SEQ + t) * HD;
#pragma unroll
    for (int i = 0; i < 16; i++) {
      const int d = dbase + i;
      const float x1 = Cs[row][d], x2 = Cs[row][d + 64];
      const float c = cosT[t * 64 + d], s = sinT[t * 64 + d];
      dst[d] = (__bf16)(x1 * c - x2 * s);
      dst[d + 64] = (__bf16)(x1 * s + x2 * c);
    }
  } else if (nb < 24) {  // K head nb-16
    __bf16* dst = kb + (((size_t)b * NKV + (nb - 16)) * T_SEQ + t) * HD;
#pragma unroll
    for (int i = 0; i < 16; i++) {
      const int d = dbase + i;
      const float x1 = Cs[row][d], x2 = Cs[row][d + 64];
      const float c = cosT[t * 64 + d], s = sinT[t * 64 + d];
      dst[d] = (__bf16)(x1 * c - x2 * s);
      dst[d + 64] = (__bf16)(x1 * s + x2 * c);
    }
  } else {  // V head nb-24, stored transposed [d][t]
    __bf16* dst = vt + ((size_t)b * NKV + (nb - 24)) * HD * T_SEQ;
#pragma unroll
    for (int i = 0; i < 16; i++) {
      const int d = dbase + i;
      dst[(size_t)d * T_SEQ + t] = (__bf16)Cs[row][d];
      dst[(size_t)(d + 64) * T_SEQ + t] = (__bf16)Cs[row][d + 64];
    }
  }
}

// ---------------------------------------------------------------------------
// Kernel 2: causal flash attention, GQA (n_rep = 2).
// One wave owns 32 query rows (two 16-row A-tiles sharing K/V B-fragments);
// kv tiles of 32; fp32 online softmax; P re-layout D->A through LDS.
// ---------------------------------------------------------------------------
__global__ __launch_bounds__(256) void flash_attn_kernel(
    const void* qb_, const void* kb_, const void* vt_, void* ob_) {
  const __bf16* qb = (const __bf16*)qb_;
  const __bf16* kb = (const __bf16*)kb_;
  const __bf16* vt = (const __bf16*)vt_;
  __bf16* ob = (__bf16*)ob_;
  __shared__ __align__(32) __bf16 P[8][32][32];

  const int tid = threadIdx.x;
  const int lane = tid & 31, wid = tid >> 5;
  const int laneHi = lane >> 4, laneLo = lane & 15;
  const int bid = blockIdx.x;
  const int qc = bid & 7;
  const int h = (bid >> 3) & 15;
  const int b = bid >> 7;
  const int hk = h >> 1;
  const int m0 = qc * 256 + wid * 32;

  const __bf16* qbase = qb + ((size_t)b * NH + h) * T_SEQ * HD;
  const __bf16* kbase = kb + ((size_t)b * NKV + hk) * T_SEQ * HD;
  const __bf16* vbase = vt + ((size_t)b * NKV + hk) * HD * T_SEQ;

  v16bf qfr[2][4];
#pragma unroll
  for (int mt = 0; mt < 2; mt++)
#pragma unroll
    for (int kk = 0; kk < 4; kk++)
      qfr[mt][kk] = load_afrag(
          qbase + (size_t)(m0 + mt * 16 + laneLo) * HD + kk * 32, laneHi);

  v8f o[2][8];
  float rmax[2][8], rsum[2][8];
#pragma unroll
  for (int mt = 0; mt < 2; mt++)
#pragma unroll
    for (int j = 0; j < 8; j++) {
      o[mt][j] = vzero(); rmax[mt][j] = -3.0e38f; rsum[mt][j] = 0.0f;
    }

  const int kend = m0 + 31;  // causal bound for this wave's 32 rows
  for (int k0 = 0; k0 <= kend; k0 += 32) {
    v8f s[2][2];
    s[0][0] = vzero(); s[0][1] = vzero(); s[1][0] = vzero(); s[1][1] = vzero();
#pragma unroll
    for (int kk = 0; kk < 4; kk++) {
      // B-frag: col (key) = lane&15, 16 contiguous d at kk*32 + 16*laneHi
      const v16bf kf0 = *(const v16bf*)(kbase + (size_t)(k0 + laneLo) * HD +
                                        kk * 32 + (laneHi << 4));
      const v16bf kf1 = *(const v16bf*)(kbase + (size_t)(k0 + 16 + laneLo) * HD +
                                        kk * 32 + (laneHi << 4));
      s[0][0] = wmma_bf16(qfr[0][kk], kf0, s[0][0]);
      s[0][1] = wmma_bf16(qfr[0][kk], kf1, s[0][1]);
      s[1][0] = wmma_bf16(qfr[1][kk], kf0, s[1][0]);
      s[1][1] = wmma_bf16(qfr[1][kk], kf1, s[1][1]);
    }
#pragma unroll
    for (int mt = 0; mt < 2; mt++) {
#pragma unroll
      for (int r = 0; r < 8; r++) {
        const int t = m0 + mt * 16 + r + (laneHi << 3);
        float v0 = s[mt][0][r] * ATT_SCALE;
        float v1 = s[mt][1][r] * ATT_SCALE;
        if (k0 + laneLo > t) v0 = -3.0e38f;
        if (k0 + 16 + laneLo > t) v1 = -3.0e38f;
        float mx = fmaxf(v0, v1);
        mx = fmaxf(mx, __shfl_xor(mx, 1, 32));
        mx = fmaxf(mx, __shfl_xor(mx, 2, 32));
        mx = fmaxf(mx, __shfl_xor(mx, 4, 32));
        mx = fmaxf(mx, __shfl_xor(mx, 8, 32));
        const float nm = fmaxf(rmax[mt][r], mx);
        const float sc = __expf(rmax[mt][r] - nm);
        const float p0 = __expf(v0 - nm);
        const float p1 = __expf(v1 - nm);
        float ps = p0 + p1;
        ps += __shfl_xor(ps, 1, 32);
        ps += __shfl_xor(ps, 2, 32);
        ps += __shfl_xor(ps, 4, 32);
        ps += __shfl_xor(ps, 8, 32);
        rsum[mt][r] = rsum[mt][r] * sc + ps;
        rmax[mt][r] = nm;
#pragma unroll
        for (int j = 0; j < 8; j++) o[mt][j][r] *= sc;  // rescale running out
        P[wid][mt * 16 + r + (laneHi << 3)][laneLo] = (__bf16)p0;
        P[wid][mt * 16 + r + (laneHi << 3)][laneLo + 16] = (__bf16)p1;
      }
    }
    asm volatile("s_wait_dscnt 0" ::: "memory");  // D-layout -> A-layout via LDS

    const v16bf pfr0 = load_afrag(&P[wid][laneLo][0], laneHi);
    const v16bf pfr1 = load_afrag(&P[wid][16 + laneLo][0], laneHi);
#pragma unroll
    for (int sub = 0; sub < 8; sub++) {
      const int d = sub * 16 + laneLo;
      // V^T layout makes the B-frag a single contiguous 32B load
      const v16bf vf =
          *(const v16bf*)(vbase + (size_t)d * T_SEQ + k0 + (laneHi << 4));
      o[0][sub] = wmma_bf16(pfr0, vf, o[0][sub]);
      o[1][sub] = wmma_bf16(pfr1, vf, o[1][sub]);
    }
  }

#pragma unroll
  for (int mt = 0; mt < 2; mt++)
#pragma unroll
    for (int r = 0; r < 8; r++) {
      const float inv = 1.0f / rsum[mt][r];
      const int t = m0 + mt * 16 + r + (laneHi << 3);
      __bf16* dst = ob + ((size_t)b * T_SEQ + t) * (NH * HD) + h * HD;
#pragma unroll
      for (int sub = 0; sub < 8; sub++)
        dst[sub * 16 + laneLo] = (__bf16)(o[mt][sub][r] * inv);
    }
}

// ---------------------------------------------------------------------------
// Kernel 3: out = attn_out @ w_o^T (M=4096, N=2048, K=2048), fp32 result.
// Block tile 128x128, 8 waves in 4x2, each wave 32x64 (8 WMMA/K-step).
// A operand (bf16) staged with GLOBAL_LOAD_ASYNC_TO_LDS_B128 (ASYNCcnt).
// ---------------------------------------------------------------------------
__global__ __launch_bounds__(256) void out_proj_kernel(
    const void* ob_, const float* __restrict__ wo, float* __restrict__ out) {
  const __bf16* ob = (const __bf16*)ob_;
  __shared__ __align__(32) __bf16 As[2][128][32];
  __shared__ __align__(32) __bf16 Bs[2][128][32];

  const int tid = threadIdx.x;
  const int lane = tid & 31, wid = tid >> 5;
  const int laneHi = lane >> 4, laneLo = lane & 15;
  const int nb = blockIdx.x, mb = blockIdx.y;
  const int m0 = mb * 128, n0 = nb * 128;
  const int wm = wid & 3, wn = wid >> 2;  // 4 x 2; wave tile 32(M) x 64(N)

  v8f acc[2][4];
#pragma unroll
  for (int i = 0; i < 2; i++)
#pragma unroll
    for (int j = 0; j < 4; j++) acc[i][j] = vzero();

  const int ldRow = tid >> 1, ldCg = (tid & 1) << 4;  // 128 rows x 32 cols

  auto stageA = [&](int buf, int k0) {
    // async DMA: bf16 tile needs no conversion -> LDS directly, ASYNCcnt
    const unsigned ldsA = (unsigned)(uintptr_t)&As[buf][ldRow][ldCg];
    const unsigned gOff =
        (unsigned)(((size_t)(m0 + ldRow) * HIDN + k0 + ldCg) * 2u);
    asm volatile("global_load_async_to_lds_b128 %0, %1, %2"
                 :: "v"(ldsA), "v"(gOff), "s"(ob)
                 : "memory");
  };
  auto stageB = [&](int buf, int k0) {
    const float4* src =
        (const float4*)(wo + (size_t)(n0 + ldRow) * HIDN + k0 + ldCg);
    __bf16* dst = &Bs[buf][ldRow][ldCg];
#pragma unroll
    for (int j = 0; j < 4; j++) {
      float4 f = src[j];
      dst[4 * j + 0] = (__bf16)f.x; dst[4 * j + 1] = (__bf16)f.y;
      dst[4 * j + 2] = (__bf16)f.z; dst[4 * j + 3] = (__bf16)f.w;
    }
  };

  stageA(0, 0);
  stageB(0, 0);
  asm volatile("s_wait_asynccnt 0" ::: "memory");
  __syncthreads();

  for (int k0 = 0; k0 < HIDN; k0 += 32) {
    const int cur = (k0 >> 5) & 1;
    if (k0 + 32 < HIDN) {
      stageA(cur ^ 1, k0 + 32);
      stageB(cur ^ 1, k0 + 32);
    }
    v16bf af0 = load_afrag(&As[cur][wm * 32 + laneLo][0], laneHi);
    v16bf af1 = load_afrag(&As[cur][wm * 32 + 16 + laneLo][0], laneHi);
    v16bf bf[4];
#pragma unroll
    for (int j = 0; j < 4; j++)
      bf[j] = *(const v16bf*)(&Bs[cur][wn * 64 + j * 16 + laneLo][laneHi << 4]);
#pragma unroll
    for (int j = 0; j < 4; j++) {
      acc[0][j] = wmma_bf16(af0, bf[j], acc[0][j]);
      acc[1][j] = wmma_bf16(af1, bf[j], acc[1][j]);
    }
    asm volatile("s_wait_asynccnt 0" ::: "memory");
    __syncthreads();
  }

#pragma unroll
  for (int i = 0; i < 2; i++)
#pragma unroll
    for (int j = 0; j < 4; j++)
#pragma unroll
      for (int r = 0; r < 8; r++) {
        const int mA = m0 + wm * 32 + i * 16 + r + (laneHi << 3);
        const int nA = n0 + wn * 64 + j * 16 + laneLo;
        out[(size_t)mA * HIDN + nA] = acc[i][j][r];
      }
}

extern "C" void kernel_launch(void* const* d_in, const int* in_sizes, int n_in,
                              void* d_out, int out_size, void* d_ws, size_t ws_size,
                              hipStream_t stream) {
  const float* hidden = (const float*)d_in[0];
  const float* cosT = (const float*)d_in[1];
  const float* sinT = (const float*)d_in[2];
  const float* wqkv = (const float*)d_in[6];
  const float* wo = (const float*)d_in[7];
  float* out = (float*)d_out;

  char* ws = (char*)d_ws;
  void* qb = ws;                                          // 16 MiB
  void* kb = ws + (size_t)16777216;                       // 8 MiB
  void* vt = ws + (size_t)16777216 + 8388608;             // 8 MiB (V transposed)
  void* ob = ws + (size_t)16777216 + 2 * (size_t)8388608; // 16 MiB

  qkv_rope_kernel<<<dim3(32, 64), 256, 0, stream>>>(hidden, cosT, sinT, wqkv,
                                                    qb, kb, vt);
  flash_attn_kernel<<<dim3(256), 256, 0, stream>>>(qb, kb, vt, ob);
  out_proj_kernel<<<dim3(16, 32), 256, 0, stream>>>(ob, wo, out);
}
